// ViTBlock_62517543961103
// MI455X (gfx1250) — matrix operand order repaired and checked by reading, not compiled, not run on hardware
//
#include <hip/hip_runtime.h>
#include <hip/hip_bf16.h>

// ---------------------------------------------------------------------------
// ViT encoder block on gfx1250 (MI455X), f16 WMMA (v_wmma_f32_16x16x32_f16)
// with f32 accumulation.
//
// Design notes (MI455X):
//  - 192MB L2: all f16 weights (~6.3MB) + activations (8MB) are L2-resident,
//    so GEMMs are LDS-free: per-lane WMMA fragments are loaded directly from
//    global (two global_load_b128 per fragment) with weights pre-transposed
//    to [N][K] so the B-fragment K-slice is contiguous.
//  - Flash attention keeps only a per-wave 16x32 P tile in LDS (C-layout ->
//    A-layout reshape); V is produced pre-transposed per head by the V
//    projection GEMM so P@V B-fragments are contiguous global loads.
//  - __builtin_prefetch -> global_prefetch_b8 for next-K-chunk streaming.
// ---------------------------------------------------------------------------

typedef _Float16 v8h  __attribute__((ext_vector_type(8)));
typedef _Float16 v16h __attribute__((ext_vector_type(16)));
typedef float    v8f  __attribute__((ext_vector_type(8)));

#define LDSS 40  // LDS row stride in halves (32 + 8 pad; 80B keeps 16B align)

__device__ __forceinline__ float gelu_exact(float x) {
    return 0.5f * x * (1.0f + erff(x * 0.70710678118654752440f));
}

__device__ __forceinline__ v8f wmma_f16(v16h a, v16h b, v8f c) {
    return __builtin_amdgcn_wmma_f32_16x16x32_f16(false, a, false, b,
                                                  (short)0, c, false, false);
}

// ---------------------------------------------------------------------------
// Weight conversion with transpose: in f32 [K][N] -> out f16 [N][K]
// ---------------------------------------------------------------------------
__global__ __launch_bounds__(256) void convT_kernel(
    const float* __restrict__ in, _Float16* __restrict__ out, int K, int N) {
    int i = blockIdx.x * 256 + threadIdx.x;
    if (i < K * N) {
        int n = i / K, k = i % K;
        out[i] = (_Float16)in[(size_t)k * N + n];
    }
}

// ---------------------------------------------------------------------------
// LayerNorm over D=512, output f16. One block (256 thr) per row.
// ---------------------------------------------------------------------------
__global__ __launch_bounds__(256) void ln_to_f16_kernel(
    const float* __restrict__ X, const float* __restrict__ g,
    const float* __restrict__ b, _Float16* __restrict__ out) {
    const int D = 512;
    const int row = blockIdx.x;
    const int t = threadIdx.x;
    const float* x = X + (size_t)row * D;

    __shared__ float sm[8];
    __shared__ float sv[8];

    float v0 = x[t], v1 = x[t + 256];
    float s = v0 + v1;
#pragma unroll
    for (int off = 1; off < 32; off <<= 1) s += __shfl_xor(s, off, 32);
    if ((t & 31) == 0) sm[t >> 5] = s;
    __syncthreads();
    float tot = 0.f;
#pragma unroll
    for (int i = 0; i < 8; ++i) tot += sm[i];
    float mu = tot * (1.0f / 512.0f);

    float d0 = v0 - mu, d1 = v1 - mu;
    float vs = d0 * d0 + d1 * d1;
#pragma unroll
    for (int off = 1; off < 32; off <<= 1) vs += __shfl_xor(vs, off, 32);
    if ((t & 31) == 0) sv[t >> 5] = vs;
    __syncthreads();
    float vtot = 0.f;
#pragma unroll
    for (int i = 0; i < 8; ++i) vtot += sv[i];
    float rstd = rsqrtf(vtot * (1.0f / 512.0f) + 1e-5f);

    out[(size_t)row * D + t]       = (_Float16)(d0 * rstd * g[t] + b[t]);
    out[(size_t)row * D + t + 256] = (_Float16)(d1 * rstd * g[t + 256] + b[t + 256]);
}

// ---------------------------------------------------------------------------
// LDS-free WMMA GEMM: C[M,N] = A[M,K] @ Bt^T, with Bt stored [N][K] f16.
// Block = 128 threads (4 wave32). Block tile 128x64; each wave owns a
// 32x64 strip: 2 m-tiles x 4 n-tiles = 8 accumulators, 8 WMMAs per K-chunk.
// Fragments are loaded straight from global; operand reuse comes from L2.
// EPI: 0 -> f16 out = acc
//      1 -> f16 out = gelu(acc + bias[n])
//      2 -> f32 out = resid + acc
//      3 -> f32 out = resid + acc + bias[n]
//      4 -> f16 out transposed per head: vT[(b*8+h)*64+dh][s]  (N=512,S=1024)
// ---------------------------------------------------------------------------
template <int EPI>
__global__ __launch_bounds__(128) void gemm_wmma(
    const _Float16* __restrict__ A, const _Float16* __restrict__ Bt,
    const float* __restrict__ bias, const float* __restrict__ resid,
    void* __restrict__ outp, int M, int N, int K) {
    const int lane = threadIdx.x & 31, wave = threadIdx.x >> 5;
    const int g = lane >> 4, ln = lane & 15;
    const int bm = blockIdx.y * 128 + wave * 32;
    const int bn = blockIdx.x * 64;

    v8f acc[2][4] = {};

    const _Float16* arow0 = A + (size_t)(bm + ln) * K;
    const _Float16* arow1 = A + (size_t)(bm + 16 + ln) * K;
    const _Float16* brow  = Bt + (size_t)(bn + ln) * K;

    for (int k0 = 0; k0 < K; k0 += 32) {
        // A fragments (contiguous K slice g*16..+15, matching B)
        v16h af[2];
        ((v8h*)&af[0])[0] = *(const v8h*)(arow0 + k0 + g * 16);
        ((v8h*)&af[0])[1] = *(const v8h*)(arow0 + k0 + g * 16 + 8);
        ((v8h*)&af[1])[0] = *(const v8h*)(arow1 + k0 + g * 16);
        ((v8h*)&af[1])[1] = *(const v8h*)(arow1 + k0 + g * 16 + 8);

        if (k0 + 32 < K) {
            __builtin_prefetch(arow0 + k0 + 32, 0, 3);
            __builtin_prefetch(arow1 + k0 + 32, 0, 3);
            __builtin_prefetch(brow + k0 + 32, 0, 3);
        }

#pragma unroll
        for (int t = 0; t < 4; ++t) {
            // B fragment: lane = column n, contiguous K slice g*16..+15
            v16h bf;
            const _Float16* p = brow + (size_t)(t * 16) * K + k0 + g * 16;
            ((v8h*)&bf)[0] = *(const v8h*)(p);
            ((v8h*)&bf)[1] = *(const v8h*)(p + 8);
            acc[0][t] = wmma_f16(af[0], bf, acc[0][t]);
            acc[1][t] = wmma_f16(af[1], bf, acc[1][t]);
        }
    }

    // Epilogue (C layout: row = r + 8*g, col = ln within n-tile)
#pragma unroll
    for (int mi = 0; mi < 2; ++mi) {
#pragma unroll
        for (int t = 0; t < 4; ++t) {
#pragma unroll
            for (int r = 0; r < 8; ++r) {
                size_t row = (size_t)(bm + mi * 16 + r + 8 * g);
                size_t col = (size_t)(bn + t * 16 + ln);
                float v = acc[mi][t][r];
                if (EPI == 1) v = gelu_exact(v + bias[col]);
                if (EPI == 2) v += resid[row * N + col];
                if (EPI == 3) v += bias[col] + resid[row * N + col];
                if (EPI == 0 || EPI == 1) {
                    ((_Float16*)outp)[row * N + col] = (_Float16)v;
                } else if (EPI == 4) {
                    // per-head transpose: N=512 (h*64+dh), S=1024
                    size_t bh = (row >> 10) * 8 + (col >> 6);
                    size_t idx = ((bh << 6) + (col & 63)) * 1024 + (row & 1023);
                    ((_Float16*)outp)[idx] = (_Float16)v;
                } else {
                    ((float*)outp)[row * N + col] = v;
                }
            }
        }
    }
}

// ---------------------------------------------------------------------------
// Flash attention: one block (128 thr / 4 waves) per (q-tile of 64, head, b).
// Q,K in f16 [B*S, D] (head slice at column h*64); V pre-transposed
// vT[(b*8+h)*64+dh][s]. Out f16 [B*S, D].
// Only LDS use: per-wave 16x32 P tile (C-layout -> A-layout reshape).
// ---------------------------------------------------------------------------
__global__ __launch_bounds__(128) void flash_attn_kernel(
    const _Float16* __restrict__ Q, const _Float16* __restrict__ Kf,
    const _Float16* __restrict__ Vt, const int* __restrict__ valid_lens,
    _Float16* __restrict__ Out, int S, int D) {
    constexpr int DH = 64;
    __shared__ __align__(16) _Float16 Plds[4][16 * LDSS];

    const int tid  = threadIdx.x;
    const int lane = tid & 31, wave = tid >> 5;
    const int g = lane >> 4, ln = lane & 15;
    const int q0 = blockIdx.x * 64;
    const int h  = blockIdx.y;
    const int b  = blockIdx.z;
    const int vl = valid_lens[b];

    const size_t baseRow = (size_t)b * S;
    const _Float16* qb  = Q + baseRow * D + h * DH;
    const _Float16* kb  = Kf + baseRow * D + h * DH;
    const _Float16* vtb = Vt + ((size_t)(b * 8 + h) * DH) * S;

    // Q fragments: 16 rows x 64 dh = 2 K-chunks of 32
    v16h aq[2];
    {
        const int qrow = q0 + wave * 16 + ln;
#pragma unroll
        for (int c = 0; c < 2; ++c) {
            const _Float16* p = qb + (size_t)qrow * D + c * 32;
            ((v8h*)&aq[c])[0] = *(const v8h*)(p + g * 16);
            ((v8h*)&aq[c])[1] = *(const v8h*)(p + g * 16 + 8);
        }
    }

    v8f o[4] = {};
    float mrun[8], lrun[8];
#pragma unroll
    for (int r = 0; r < 8; ++r) { mrun[r] = -3.0e38f; lrun[r] = 0.0f; }
    const float scale = 0.125f;  // 1/sqrt(64)

    for (int k0 = 0; k0 < S; k0 += 32) {
        // scores 16x32: Q @ K^T  (row-major K row == B-fragment column)
        v8f sc[2] = {};
#pragma unroll
        for (int nt = 0; nt < 2; ++nt) {
#pragma unroll
            for (int c = 0; c < 2; ++c) {
                v16h bf;
                const _Float16* p =
                    kb + (size_t)(k0 + nt * 16 + ln) * D + c * 32 + g * 16;
                ((v8h*)&bf)[0] = *(const v8h*)(p);
                ((v8h*)&bf)[1] = *(const v8h*)(p + 8);
                sc[nt] = wmma_f16(aq[c], bf, sc[nt]);
            }
        }
        if (k0 + 32 < S) {
            __builtin_prefetch(kb + (size_t)(k0 + 32 + ln) * D, 0, 3);
            __builtin_prefetch(vtb + (size_t)(ln)*S + k0 + 32, 0, 3);
        }

        // scale + key mask (column index is per-lane uniform over rows)
        float sv[2][8];
#pragma unroll
        for (int nt = 0; nt < 2; ++nt) {
            const int key = k0 + nt * 16 + ln;
            const bool masked = key >= vl;
#pragma unroll
            for (int r = 0; r < 8; ++r)
                sv[nt][r] = masked ? -1.0e6f : sc[nt][r] * scale;
        }

        // online softmax per row r within each 16-lane group
#pragma unroll
        for (int r = 0; r < 8; ++r) {
            float mx = fmaxf(sv[0][r], sv[1][r]);
#pragma unroll
            for (int off = 1; off < 16; off <<= 1)
                mx = fmaxf(mx, __shfl_xor(mx, off, 32));
            float mnew  = fmaxf(mrun[r], mx);
            float alpha = __expf(mrun[r] - mnew);
            float p0 = __expf(sv[0][r] - mnew);
            float p1 = __expf(sv[1][r] - mnew);
            sv[0][r] = p0; sv[1][r] = p1;
            float ls = p0 + p1;
#pragma unroll
            for (int off = 1; off < 16; off <<= 1)
                ls += __shfl_xor(ls, off, 32);
            lrun[r] = lrun[r] * alpha + ls;
            mrun[r] = mnew;
#pragma unroll
            for (int t = 0; t < 4; ++t) o[t][r] *= alpha;
        }

        // P (C layout) -> per-wave LDS -> A layout fragment
        _Float16* pl = Plds[wave];
#pragma unroll
        for (int nt = 0; nt < 2; ++nt)
#pragma unroll
            for (int r = 0; r < 8; ++r)
                pl[(r + 8 * g) * LDSS + nt * 16 + ln] = (_Float16)sv[nt][r];
        __syncthreads();

        v16h ap;
        {
            const _Float16* p = pl + ln * LDSS;
            ((v8h*)&ap)[0] = *(const v8h*)(p + g * 16);
            ((v8h*)&ap)[1] = *(const v8h*)(p + g * 16 + 8);
        }
#pragma unroll
        for (int t = 0; t < 4; ++t) {
            // P@V B fragment: column = dh, contiguous keys from vT
            v16h bf;
            const _Float16* p = vtb + (size_t)(t * 16 + ln) * S + k0 + g * 16;
            ((v8h*)&bf)[0] = *(const v8h*)(p);
            ((v8h*)&bf)[1] = *(const v8h*)(p + 8);
            o[t] = wmma_f16(ap, bf, o[t]);
        }
        __syncthreads();  // Plds reuse next iteration
    }

    // normalize and store to [B*S, D] at column h*64
#pragma unroll
    for (int r = 0; r < 8; ++r) {
        float inv = 1.0f / lrun[r];
        size_t row = baseRow + (size_t)(q0 + wave * 16 + r + 8 * g);
#pragma unroll
        for (int t = 0; t < 4; ++t) {
            int col = h * DH + t * 16 + ln;
            Out[row * D + col] = (_Float16)(o[t][r] * inv);
        }
    }
}

// ---------------------------------------------------------------------------
// Host launcher
// ---------------------------------------------------------------------------
extern "C" void kernel_launch(void* const* d_in, const int* in_sizes, int n_in,
                              void* d_out, int out_size, void* d_ws, size_t ws_size,
                              hipStream_t stream) {
    (void)in_sizes; (void)n_in; (void)out_size; (void)ws_size;
    constexpr int B = 8, S = 1024, D = 512, F = 2048;
    constexpr int M = B * S;  // 8192

    const float* X    = (const float*)d_in[0];
    const int*   vln  = (const int*)d_in[1];
    const float* ln1g = (const float*)d_in[2];
    const float* ln1b = (const float*)d_in[3];
    const float* Wq = (const float*)d_in[4];
    const float* Wk = (const float*)d_in[5];
    const float* Wv = (const float*)d_in[6];
    const float* Wo = (const float*)d_in[7];
    const float* ln2g = (const float*)d_in[8];
    const float* ln2b = (const float*)d_in[9];
    const float* W1 = (const float*)d_in[10];
    const float* b1 = (const float*)d_in[11];
    const float* W2 = (const float*)d_in[12];
    const float* b2 = (const float*)d_in[13];
    float* out = (float*)d_out;

    // Workspace carve-out
    char* w = (char*)d_ws;
    auto alloc = [&](size_t bytes) -> char* {
        char* p = w;
        w += (bytes + 255) & ~(size_t)255;
        return p;
    };
    _Float16* h16   = (_Float16*)alloc((size_t)M * D * 2);
    _Float16* WqT   = (_Float16*)alloc((size_t)D * D * 2);
    _Float16* WkT   = (_Float16*)alloc((size_t)D * D * 2);
    _Float16* WvT   = (_Float16*)alloc((size_t)D * D * 2);
    _Float16* WoT   = (_Float16*)alloc((size_t)D * D * 2);
    _Float16* W1T   = (_Float16*)alloc((size_t)D * F * 2);  // [F][D]
    _Float16* W2T   = (_Float16*)alloc((size_t)F * D * 2);  // [D][F]
    _Float16* q16   = (_Float16*)alloc((size_t)M * D * 2);
    _Float16* k16   = (_Float16*)alloc((size_t)M * D * 2);
    _Float16* vT16  = (_Float16*)alloc((size_t)M * D * 2);  // [B*H*64][S]
    _Float16* a16   = (_Float16*)alloc((size_t)M * D * 2);
    float*    X2    = (float*)alloc((size_t)M * D * 4);
    _Float16* h216  = (_Float16*)alloc((size_t)M * D * 2);
    _Float16* m116  = (_Float16*)alloc((size_t)M * F * 2);

    // 1) weight conversion f32 [K][N] -> f16 transposed [N][K]
    auto convT = [&](const float* src, _Float16* dst, int K, int N) {
        int n = K * N;
        convT_kernel<<<(n + 255) / 256, 256, 0, stream>>>(src, dst, K, N);
    };
    convT(Wq, WqT, D, D);
    convT(Wk, WkT, D, D);
    convT(Wv, WvT, D, D);
    convT(Wo, WoT, D, D);
    convT(W1, W1T, D, F);
    convT(W2, W2T, F, D);

    // 2) LN1(X) -> h16
    ln_to_f16_kernel<<<M, 256, 0, stream>>>(X, ln1g, ln1b, h16);

    // 3) QKV projections (f16 WMMA, LDS-free); V written pre-transposed
    dim3 blk(128);
    dim3 gProj(D / 64, M / 128);
    gemm_wmma<0><<<gProj, blk, 0, stream>>>(h16, WqT, nullptr, nullptr, q16, M, D, D);
    gemm_wmma<0><<<gProj, blk, 0, stream>>>(h16, WkT, nullptr, nullptr, k16, M, D, D);
    gemm_wmma<4><<<gProj, blk, 0, stream>>>(h16, WvT, nullptr, nullptr, vT16, M, D, D);

    // 4) flash attention -> a16
    dim3 gAttn(S / 64, 8 /*H*/, B);
    flash_attn_kernel<<<gAttn, blk, 0, stream>>>(q16, k16, vT16, vln, a16, S, D);

    // 5) X2 = X + a16 @ Wo   (f32 residual)
    gemm_wmma<2><<<gProj, blk, 0, stream>>>(a16, WoT, nullptr, X, X2, M, D, D);

    // 6) LN2(X2) -> h216
    ln_to_f16_kernel<<<M, 256, 0, stream>>>(X2, ln2g, ln2b, h216);

    // 7) m116 = gelu(h216 @ W1 + b1)
    dim3 gMlp1(F / 64, M / 128);
    gemm_wmma<1><<<gMlp1, blk, 0, stream>>>(h216, W1T, b1, nullptr, m116, M, F, D);

    // 8) out = X2 + m116 @ W2 + b2
    gemm_wmma<3><<<gProj, blk, 0, stream>>>(m116, W2T, b2, X2, out, M, D, F);
}
